// QNetNode_88416196755624
// MI455X (gfx1250) — matrix operand order repaired and checked by reading, not compiled, not run on hardware
//
#include <hip/hip_runtime.h>

// ---------------------------------------------------------------------------
// QNet GNN pipeline for MI455X (gfx1250, wave32, WMMA).
//   conv1 : scatter-mean over 800k edges -> fused [N,256]x[256,64] bf16 WMMA
//   conv2 : per-graph, compacted to the <=2001 needed nodes (regions+target)
//   head  : q[a] = relu(h2[a]@Wm+bm) . (Wo@tgt)  + bo.tgt   (2nd GEMM folded)
// B-matrix weights are staged in LDS pre-swizzled into WMMA fragment order so
// each lane loads its 16 bf16 values with one aligned 32-byte DS access.
// ---------------------------------------------------------------------------

#define NN   50000
#define E1N  800000
#define E2N  200000
#define BN   16
#define AN   2000
#define SLOTS 2016          // 2001 needed rows padded to 16-row tiles

typedef __attribute__((ext_vector_type(16))) __bf16 v16bf;
typedef __attribute__((ext_vector_type(8)))  float  v8f;

// A-fragment for v_wmma_f32_16x16x32_bf16 (ISA 7.12.2, 16-bit A 16x32):
// lane L holds row M=L&15; lanes 0-15 carry K-octets {0..7,16..23},
// lanes 16-31 carry {8..15,24..31}.  row32 = &A[M][ktile*32], scaled.
__device__ __forceinline__ v16bf load_a_frag(const float* row32, float scale, int lane) {
    const int hb = (lane >> 4) << 1;            // float4 index offset: 0 or 2
    const float4* p = (const float4*)row32;
    float4 q0 = p[hb + 0];
    float4 q1 = p[hb + 1];
    float4 q2 = p[hb + 4];
    float4 q3 = p[hb + 5];
    v16bf a;
    a[0]=(__bf16)(q0.x*scale); a[1]=(__bf16)(q0.y*scale); a[2]=(__bf16)(q0.z*scale); a[3]=(__bf16)(q0.w*scale);
    a[4]=(__bf16)(q1.x*scale); a[5]=(__bf16)(q1.y*scale); a[6]=(__bf16)(q1.z*scale); a[7]=(__bf16)(q1.w*scale);
    a[8]=(__bf16)(q2.x*scale); a[9]=(__bf16)(q2.y*scale); a[10]=(__bf16)(q2.z*scale); a[11]=(__bf16)(q2.w*scale);
    a[12]=(__bf16)(q3.x*scale); a[13]=(__bf16)(q3.y*scale); a[14]=(__bf16)(q3.z*scale); a[15]=(__bf16)(q3.w*scale);
    return a;
}

// B fragment from LDS pre-swizzled storage: one contiguous 32B read per lane.
__device__ __forceinline__ v16bf load_b_frag_sw(const v16bf* sBf, int fragIdx, int lane) {
    return sBf[fragIdx * 32 + lane];
}

// Fill pre-swizzled B storage: dest element i -> (frag f, lane L, elem j) with
// K = kt*32 + ((j&8)<<1) + ((L>>4)<<3) + (j&7), Nc = nt*16 + (L&15); f = kt*NT + nt.
// getw(k,n) supplies the f32 weight.
template <typename GETW>
__device__ __forceinline__ void fill_b_swizzled(__bf16* raw, int KT, int NT, int tid,
                                                int nthreads, GETW getw) {
    const int total = KT * NT * 32 * 16;
    for (int i = tid; i < total; i += nthreads) {
        int j    = i & 15;
        int lane = (i >> 4) & 31;
        int f    = i >> 9;
        int nt   = f % NT;
        int kt   = f / NT;
        int k    = kt * 32 + ((j & 8) << 1) + ((lane >> 4) << 3) + (j & 7);
        int n    = nt * 16 + (lane & 15);
        raw[i] = (__bf16)getw(k, n);
    }
}

// ------------------------------- fills -------------------------------------
__global__ void fill_f32(float* p, float v, int n) {
    int i = blockIdx.x * blockDim.x + threadIdx.x;
    if (i < n) p[i] = v;
}
__global__ void fill_i32(int* p, int v, int n) {
    int i = blockIdx.x * blockDim.x + threadIdx.x;
    if (i < n) p[i] = v;
}

// ------------------------- conv1 scatter (mean agg) ------------------------
// one wave per edge; lane handles 4 of 128 features (float4 gather).
__global__ void scatter1_kernel(const float* __restrict__ x, const int* __restrict__ ei,
                                float* __restrict__ agg, float* __restrict__ deg) {
    long gid = (long)blockIdx.x * blockDim.x + threadIdx.x;
    int e = (int)(gid >> 5), lane = (int)(gid & 31);
    if (e >= E1N) return;
    int s = ei[e];
    int d = ei[E1N + e];
    float4 v = ((const float4*)(x + (size_t)s * 128))[lane];
    float* ap = agg + (size_t)d * 128 + lane * 4;
    atomicAdd(ap + 0, v.x); atomicAdd(ap + 1, v.y);
    atomicAdd(ap + 2, v.z); atomicAdd(ap + 3, v.w);
    if (lane == 0) atomicAdd(deg + d, 1.0f);
}

// ------------------- conv1 GEMM: h1 = relu([mean|x]@[W1l;W1r]+b1) ----------
__global__ void __launch_bounds__(256)
gemm1_kernel(const float* __restrict__ agg1, const float* __restrict__ deg1,
             const float* __restrict__ x, const float* __restrict__ W1l,
             const float* __restrict__ W1r, const float* __restrict__ b1,
             float* __restrict__ h1) {
    __shared__ v16bf sBf[8 * 4 * 32];                     // 32 KB, frag order
    int tid = threadIdx.x;
    fill_b_swizzled((__bf16*)sBf, 8, 4, tid, 256, [&](int k, int n) {
        return (k < 128) ? W1l[k * 64 + n] : W1r[(k - 128) * 64 + n];
    });
    __syncthreads();

    int wid = tid >> 5, lane = tid & 31;
    int t = blockIdx.x * 8 + wid;                         // 16-row node tile
    if (t >= NN / 16) return;                             // uniform per wave
    int col = lane & 15, hi = lane >> 4;
    int node = t * 16 + col;                              // lane's A-row
    float rd = 1.0f / fmaxf(deg1[node], 1.0f);

    v8f c[4] = {};
#pragma unroll
    for (int kt = 0; kt < 8; ++kt) {                      // K = 256
        const float* row32 = (kt < 4) ? (agg1 + (size_t)node * 128 + kt * 32)
                                      : (x    + (size_t)node * 128 + (kt - 4) * 32);
        float sc = (kt < 4) ? rd : 1.0f;
        v16bf a = load_a_frag(row32, sc, lane);
#pragma unroll
        for (int nt = 0; nt < 4; ++nt) {
            v16bf bf = load_b_frag_sw(sBf, kt * 4 + nt, lane);
            c[nt] = __builtin_amdgcn_wmma_f32_16x16x32_bf16(
                false, a, false, bf, (short)0, c[nt], false, false);
        }
    }
#pragma unroll
    for (int nt = 0; nt < 4; ++nt) {
        float bb = b1[nt * 16 + col];
#pragma unroll
        for (int j = 0; j < 8; ++j) {
            int row = t * 16 + j + hi * 8;
            h1[(size_t)row * 64 + nt * 16 + col] = fmaxf(c[nt][j] + bb, 0.0f);
        }
    }
}

// ---------------- per-graph: mark needed nodes (dedup via atomicMin) -------
__global__ void mark_kernel(const int* __restrict__ region, const int* __restrict__ tgt,
                            int* __restrict__ flags, int b) {
    int a = blockIdx.x * blockDim.x + threadIdx.x;
    if (a < AN)       atomicMin(&flags[region[b * AN + a]], a);
    else if (a == AN) atomicMin(&flags[tgt[b]], AN);
}

// --------- per-graph scatter2: only edges into needed nodes (compact) ------
__global__ void scatter2_kernel(const float* __restrict__ h1, const int* __restrict__ ne,
                                const int* __restrict__ flags, float* __restrict__ cagg,
                                float* __restrict__ cdeg, int b) {
    long gid = (long)blockIdx.x * blockDim.x + threadIdx.x;
    int e = (int)(gid >> 5), lane = (int)(gid & 31);
    if (e >= E2N) return;
    int d = ne[(size_t)b * 2 * E2N + E2N + e];
    int slot = flags[d];
    if (slot > AN) return;                                 // not a needed dst
    int s = ne[(size_t)b * 2 * E2N + e];
    float2 v = ((const float2*)(h1 + (size_t)s * 64))[lane];
    float* cp = cagg + (size_t)slot * 64 + lane * 2;
    atomicAdd(cp, v.x); atomicAdd(cp + 1, v.y);
    if (lane == 0) atomicAdd(cdeg + slot, 1.0f);
}

// ------- conv2 GEMM on 2016 compact rows: h2 = relu([mean2|h1]@[W2l;W2r]+b2)
__global__ void __launch_bounds__(256)
gemm2_kernel(const float* __restrict__ cagg, const float* __restrict__ cdeg,
             const float* __restrict__ h1, const int* __restrict__ flags,
             const int* __restrict__ region, const int* __restrict__ tgt,
             const float* __restrict__ W2l, const float* __restrict__ W2r,
             const float* __restrict__ b2, float* __restrict__ h2buf, int b) {
    __shared__ v16bf sBf[4 * 4 * 32];                     // 16 KB, frag order
    int tid = threadIdx.x;
    fill_b_swizzled((__bf16*)sBf, 4, 4, tid, 256, [&](int k, int n) {
        return (k < 64) ? W2l[k * 64 + n] : W2r[(k - 64) * 64 + n];
    });
    __syncthreads();

    int wid = tid >> 5, lane = tid & 31;
    int t = blockIdx.x * 8 + wid;
    if (t >= SLOTS / 16) return;
    int col = lane & 15, hi = lane >> 4;
    int sidx = t * 16 + col; if (sidx > AN) sidx = AN;     // pad rows dup target
    int v  = (sidx < AN) ? region[b * AN + sidx] : tgt[b];
    int cs = flags[v];                                     // canonical slot
    float rd = 1.0f / fmaxf(cdeg[cs], 1.0f);

    v8f c[4] = {};
#pragma unroll
    for (int kt = 0; kt < 4; ++kt) {                       // K = 128
        const float* row32 = (kt < 2) ? (cagg + (size_t)cs * 64 + kt * 32)
                                      : (h1  + (size_t)v  * 64 + (kt - 2) * 32);
        float sc = (kt < 2) ? rd : 1.0f;
        v16bf a = load_a_frag(row32, sc, lane);
#pragma unroll
        for (int nt = 0; nt < 4; ++nt) {
            v16bf bf = load_b_frag_sw(sBf, kt * 4 + nt, lane);
            c[nt] = __builtin_amdgcn_wmma_f32_16x16x32_bf16(
                false, a, false, bf, (short)0, c[nt], false, false);
        }
    }
#pragma unroll
    for (int nt = 0; nt < 4; ++nt) {
        float bb = b2[nt * 16 + col];
#pragma unroll
        for (int j = 0; j < 8; ++j) {
            int row = t * 16 + j + hi * 8;
            h2buf[(size_t)row * 64 + nt * 16 + col] = fmaxf(c[nt][j] + bb, 0.0f);
        }
    }
}

// ------- per-graph: wv = Wo @ target_embed, wv[128] = bo . target_embed ----
__global__ void wv_kernel(const float* __restrict__ Wo, const float* __restrict__ bo,
                          const float* __restrict__ h2buf, float* __restrict__ wv) {
    const float* tg = h2buf + (size_t)AN * 64;            // row 2000 = target
    int h = threadIdx.x;
    if (h < 128) {
        float s = 0.0f;
        for (int d = 0; d < 64; ++d) s += Wo[h * 64 + d] * tg[d];
        wv[h] = s;
    } else if (h == 128) {
        float s = 0.0f;
        for (int d = 0; d < 64; ++d) s += bo[d] * tg[d];
        wv[128] = s;
    }
}

// ---- head: q[a] = sum_h relu(h2[a]@Wm + bm)[h] * wv[h] + c0  (WMMA GEMM) --
__global__ void __launch_bounds__(256)
qhead_kernel(const float* __restrict__ h2buf, const float* __restrict__ Wm,
             const float* __restrict__ bm, const float* __restrict__ wv,
             float* __restrict__ out, int b) {
    __shared__ v16bf sBf[2 * 8 * 32];                     // 16 KB: Wm frag order
    int tid = threadIdx.x;
    fill_b_swizzled((__bf16*)sBf, 2, 8, tid, 256, [&](int k, int n) {
        return Wm[k * 128 + n];
    });
    __syncthreads();

    int wid = tid >> 5, lane = tid & 31;
    int t = blockIdx.x * 8 + wid;
    if (t >= AN / 16) return;                              // 125 tiles
    int col = lane & 15, hi = lane >> 4;
    int row = t * 16 + col;

    v8f c[8] = {};
#pragma unroll
    for (int kt = 0; kt < 2; ++kt) {                       // K = 64
        v16bf a = load_a_frag(h2buf + (size_t)row * 64 + kt * 32, 1.0f, lane);
#pragma unroll
        for (int nt = 0; nt < 8; ++nt) {                   // N = 128
            v16bf bf = load_b_frag_sw(sBf, kt * 8 + nt, lane);
            c[nt] = __builtin_amdgcn_wmma_f32_16x16x32_bf16(
                false, a, false, bf, (short)0, c[nt], false, false);
        }
    }
    float part[8] = {0.f, 0.f, 0.f, 0.f, 0.f, 0.f, 0.f, 0.f};
#pragma unroll
    for (int nt = 0; nt < 8; ++nt) {
        float wvn = wv[nt * 16 + col];
        float bmn = bm[nt * 16 + col];
#pragma unroll
        for (int j = 0; j < 8; ++j)
            part[j] += fmaxf(c[nt][j] + bmn, 0.0f) * wvn;
    }
#pragma unroll
    for (int off = 1; off < 16; off <<= 1)                 // reduce across N lanes
#pragma unroll
        for (int j = 0; j < 8; ++j)
            part[j] += __shfl_xor(part[j], off, 32);
    if (col == 0) {
        float c0 = wv[128];
#pragma unroll
        for (int j = 0; j < 8; ++j)
            out[b * AN + t * 16 + j + hi * 8] = part[j] + c0;
    }
}

// ---------------------------------------------------------------------------
extern "C" void kernel_launch(void* const* d_in, const int* in_sizes, int n_in,
                              void* d_out, int out_size, void* d_ws, size_t ws_size,
                              hipStream_t stream) {
    (void)in_sizes; (void)n_in; (void)out_size; (void)ws_size;
    const float* x   = (const float*)d_in[0];
    const int*   ei  = (const int*)d_in[1];
    const int*   ne  = (const int*)d_in[2];
    const int*   tg  = (const int*)d_in[3];
    const int*   rg  = (const int*)d_in[4];
    const float* W1l = (const float*)d_in[5];
    const float* W1r = (const float*)d_in[6];
    const float* b1  = (const float*)d_in[7];
    const float* W2l = (const float*)d_in[8];
    const float* W2r = (const float*)d_in[9];
    const float* b2  = (const float*)d_in[10];
    const float* Wm  = (const float*)d_in[11];
    const float* bm  = (const float*)d_in[12];
    const float* Wo  = (const float*)d_in[13];
    const float* bo  = (const float*)d_in[14];
    float* out = (float*)d_out;

    // workspace layout (all f32-sized slots, ~39.9 MB, 16B-aligned pieces)
    float* agg1  = (float*)d_ws;                 // [N,128]
    float* deg1  = agg1 + (size_t)NN * 128;      // [N]
    float* h1    = deg1 + NN;                    // [N,64]
    int*   flags = (int*)(h1 + (size_t)NN * 64); // [N]
    float* cagg  = (float*)(flags + NN);         // [SLOTS,64]
    float* cdeg  = cagg + (size_t)SLOTS * 64;    // [SLOTS]
    float* h2buf = cdeg + SLOTS;                 // [SLOTS,64]
    float* wv    = h2buf + (size_t)SLOTS * 64;   // [132]

    // conv1: zero accumulators, scatter-mean, fused GEMM
    {
        int n = NN * 129;                        // agg1+deg1 contiguous
        fill_f32<<<(n + 255) / 256, 256, 0, stream>>>(agg1, 0.0f, n);
        long th = (long)E1N * 32;
        scatter1_kernel<<<(unsigned)((th + 255) / 256), 256, 0, stream>>>(x, ei, agg1, deg1);
        gemm1_kernel<<<(NN / 16 + 7) / 8, 256, 0, stream>>>(agg1, deg1, x, W1l, W1r, b1, h1);
    }

    for (int b = 0; b < BN; ++b) {
        fill_i32<<<(NN + 255) / 256, 256, 0, stream>>>(flags, 0x7FFFFFFF, NN);
        int nz = SLOTS * 64 + SLOTS;             // cagg+cdeg contiguous
        fill_f32<<<(nz + 255) / 256, 256, 0, stream>>>(cagg, 0.0f, nz);
        mark_kernel<<<(AN + 1 + 255) / 256, 256, 0, stream>>>(rg, tg, flags, b);
        long th = (long)E2N * 32;
        scatter2_kernel<<<(unsigned)((th + 255) / 256), 256, 0, stream>>>(h1, ne, flags, cagg, cdeg, b);
        gemm2_kernel<<<(SLOTS / 16 + 7) / 8, 256, 0, stream>>>(
            cagg, cdeg, h1, flags, rg, tg, W2l, W2r, b2, h2buf, b);
        wv_kernel<<<1, 160, 0, stream>>>(Wo, bo, h2buf, wv);
        qhead_kernel<<<(AN / 16 + 7) / 8, 256, 0, stream>>>(h2buf, Wm, bm, wv, out, b);
    }
}
